// VisionTransformerTokenPruningV1_36971078484205
// MI455X (gfx1250) — compile-verified, matrix operand order checked by reading, not run on hardware
//
#include <hip/hip_runtime.h>
#include <hip/hip_bf16.h>

#define BB 32
#define DD 768
#define NHH 12
#define HDD 64

typedef __attribute__((ext_vector_type(16))) _Float16 v16h;
typedef __attribute__((ext_vector_type(8)))  _Float16 v8h;
typedef __attribute__((ext_vector_type(8)))  float    v8f;

// ---------------------------------------------------------------- fragments
// 16-bit A/B fragment (16x32 / 32x16): per-lane K pattern is two contiguous
// 8-half runs:  K = (e/8)*16 + (lane>=16)*8 + e%8   (ISA 7.12.2)
__device__ __forceinline__ v16h load_frag16(const _Float16* rowk, int hi) {
  const _Float16* p = rowk + hi * 8;
  v8h a = *(const v8h*)(p);
  v8h b = *(const v8h*)(p + 16);
  v16h r;
#pragma unroll
  for (int i = 0; i < 8; ++i) { r[i] = a[i]; r[i + 8] = b[i]; }
  return r;
}

__device__ __forceinline__ float gelu_exact(float x) {
  return 0.5f * x * (1.0f + erff(x * 0.70710678118654752f));
}

// ---------------------------------------------------------------- direct GEMM
// C[M,N] = act( A[M,K] * W[N,K]^T + bias + residual ), K % 32 == 0
// block = 256 thr = 8 waves; block tile 64(M) x 128(N); wave tile 16 x 64.
// Used for the small-K attention GEMMs.
__global__ __launch_bounds__(256) void gemm_wmma(
    const _Float16* __restrict__ A, long long sA,
    const _Float16* __restrict__ W, long long sW,
    const float* __restrict__ bias,
    const float* __restrict__ Res, long long sR,
    float* __restrict__ Cf, long long sCf,
    _Float16* __restrict__ Ch, long long sCh,
    int M, int N, int K, int act)
{
  const int z = blockIdx.z;
  A += (long long)z * sA;
  W += (long long)z * sW;

  const int lane = threadIdx.x & 31;
  const int wave = threadIdx.x >> 5;
  const int wm = wave & 3;
  const int wn = wave >> 2;
  const int hi  = lane >> 4;
  const int l16 = lane & 15;

  const int m0 = blockIdx.x * 64 + wm * 16;
  const int n0 = blockIdx.y * 128 + wn * 64;

  int ra = m0 + l16; if (ra > M - 1) ra = M - 1;
  const _Float16* aRow = A + (long long)ra * K;

  const _Float16* wRow[4];
#pragma unroll
  for (int t = 0; t < 4; ++t) {
    int rw = n0 + t * 16 + l16; if (rw > N - 1) rw = N - 1;
    wRow[t] = W + (long long)rw * K;
  }

  v8f acc[4] = {};

  for (int k0 = 0; k0 < K; k0 += 32) {
    __builtin_prefetch(aRow + k0 + 256, 0, 1);
    __builtin_prefetch(wRow[0] + k0 + 256, 0, 1);
    v16h af = load_frag16(aRow + k0, hi);
#pragma unroll
    for (int t = 0; t < 4; ++t) {
      v16h bf = load_frag16(wRow[t] + k0, hi);
      acc[t] = __builtin_amdgcn_wmma_f32_16x16x32_f16(
          false, af, false, bf, (short)0, acc[t], false, false);
    }
  }

  const float* res = Res ? Res + (long long)z * sR : nullptr;
  float*       cf  = Cf  ? Cf  + (long long)z * sCf : nullptr;
  _Float16*    ch  = Ch  ? Ch  + (long long)z * sCh : nullptr;

#pragma unroll
  for (int t = 0; t < 4; ++t) {
    int n = n0 + t * 16 + l16;
    if (n >= N) continue;
    float bv = bias ? bias[n] : 0.0f;
#pragma unroll
    for (int r = 0; r < 8; ++r) {
      int m = m0 + r + 8 * hi;       // C layout: VGPR r -> M = r + 8*(lane/16)
      if (m >= M) continue;
      float v = acc[t][r] + bv;
      long long off = (long long)m * N + n;
      if (res) v += res[off];
      if (act == 1) v = gelu_exact(v);
      if (cf) cf[off] = v;
      if (ch) ch[off] = (_Float16)v;
    }
  }
}

// ------------------------------------------------- async-LDS staged GEMM
// Same math as gemm_wmma, but the 128x32 W tile of each K-chunk is staged
// into LDS once per block via gfx1250 async global->LDS copies
// (global_load_async_to_lds_b128, ASYNCcnt, s_wait_asynccnt), double
// buffered; B fragments then come from LDS (ds_load_b128). W rows padded
// to 40 halves to spread LDS banks. Requires K % 32 == 0.
#define SW_STRIDE 40
#define SW_BUFSZ  (128 * SW_STRIDE)

__device__ __forceinline__ void stage_w_chunk(const _Float16* __restrict__ W,
                                              int N, int K, int n0b, int k0,
                                              _Float16* swbase) {
  const int t = threadIdx.x;
#pragma unroll
  for (int j = 0; j < 2; ++j) {
    int seg = t + j * 256;          // 512 segs = 128 rows x 4 x 16B
    int r = seg >> 2;
    int s4 = seg & 3;
    int rw = n0b + r; if (rw > N - 1) rw = N - 1;
    const _Float16* g = W + (long long)rw * K + k0 + s4 * 8;
    unsigned lo = (unsigned)(size_t)(swbase + r * SW_STRIDE + s4 * 8);
    asm volatile("global_load_async_to_lds_b128 %0, %1, off"
                 :: "v"(lo), "v"(g) : "memory");
  }
}

__global__ __launch_bounds__(256) void gemm_wmma_lds(
    const _Float16* __restrict__ A, long long sA,
    const _Float16* __restrict__ W, long long sW,
    const float* __restrict__ bias,
    const float* __restrict__ Res, long long sR,
    float* __restrict__ Cf, long long sCf,
    _Float16* __restrict__ Ch, long long sCh,
    int M, int N, int K, int act)
{
  __shared__ __align__(16) _Float16 sw[2 * SW_BUFSZ];

  const int z = blockIdx.z;
  A += (long long)z * sA;
  W += (long long)z * sW;

  const int lane = threadIdx.x & 31;
  const int wave = threadIdx.x >> 5;
  const int wm = wave & 3;
  const int wn = wave >> 2;
  const int hi  = lane >> 4;
  const int l16 = lane & 15;

  const int m0 = blockIdx.x * 64 + wm * 16;
  const int n0b = blockIdx.y * 128;          // block N base (LDS tile base)
  const int nloc = wn * 64;                  // this wave's offset in the tile

  int ra = m0 + l16; if (ra > M - 1) ra = M - 1;
  const _Float16* aRow = A + (long long)ra * K;

  v8f acc[4] = {};

  const int nch = K >> 5;
  stage_w_chunk(W, N, K, n0b, 0, sw);

  for (int kc = 0; kc < nch; ++kc) {
    const int buf = kc & 1;
    if (kc + 1 < nch) {
      stage_w_chunk(W, N, K, n0b, (kc + 1) << 5, sw + (buf ^ 1) * SW_BUFSZ);
      asm volatile("s_wait_asynccnt 0x2" ::: "memory");
    } else {
      asm volatile("s_wait_asynccnt 0x0" ::: "memory");
    }
    __syncthreads();

    const int k0 = kc << 5;
    __builtin_prefetch(aRow + k0 + 256, 0, 1);
    v16h af = load_frag16(aRow + k0, hi);
    const _Float16* sb = sw + buf * SW_BUFSZ;
#pragma unroll
    for (int t = 0; t < 4; ++t) {
      const _Float16* p = sb + (nloc + t * 16 + l16) * SW_STRIDE + hi * 8;
      v8h x0 = *(const v8h*)(p);
      v8h x1 = *(const v8h*)(p + 16);
      v16h bf;
#pragma unroll
      for (int i = 0; i < 8; ++i) { bf[i] = x0[i]; bf[i + 8] = x1[i]; }
      acc[t] = __builtin_amdgcn_wmma_f32_16x16x32_f16(
          false, af, false, bf, (short)0, acc[t], false, false);
    }
    __syncthreads();   // all reads of this buffer done before it is re-staged
  }

  const float* res = Res ? Res + (long long)z * sR : nullptr;
  float*       cf  = Cf  ? Cf  + (long long)z * sCf : nullptr;
  _Float16*    ch  = Ch  ? Ch  + (long long)z * sCh : nullptr;

#pragma unroll
  for (int t = 0; t < 4; ++t) {
    int n = n0b + nloc + t * 16 + l16;
    if (n >= N) continue;
    float bv = bias ? bias[n] : 0.0f;
#pragma unroll
    for (int r = 0; r < 8; ++r) {
      int m = m0 + r + 8 * hi;
      if (m >= M) continue;
      float v = acc[t][r] + bv;
      long long off = (long long)m * N + n;
      if (res) v += res[off];
      if (act == 1) v = gelu_exact(v);
      if (cf) cf[off] = v;
      if (ch) ch[off] = (_Float16)v;
    }
  }
}

// ---------------------------------------------------------------- elementwise
__global__ void f32_to_f16(const float* __restrict__ in, _Float16* __restrict__ out,
                           long long n) {
  long long i = (long long)blockIdx.x * blockDim.x + threadIdx.x;
  if (i < n) out[i] = (_Float16)in[i];
}

__global__ void im2col_kernel(const float* __restrict__ x, _Float16* __restrict__ P,
                              long long total) {
  long long gid = (long long)blockIdx.x * blockDim.x + threadIdx.x;
  if (gid >= total) return;
  int c = (int)(gid % DD);
  long long row = gid / DD;
  int b = (int)(row / 196), l = (int)(row % 196);
  int gy = l / 14, gx = l % 14;
  int ch = c / 256, rem = c % 256, py = rem / 16, px = rem % 16;
  P[gid] = (_Float16)x[(((long long)(b * 3 + ch)) * 224 + gy * 16 + py) * 224 +
                       gx * 16 + px];
}

__global__ void add_clspos(float* __restrict__ t, const float* __restrict__ cls,
                           const float* __restrict__ pos, long long total) {
  long long gid = (long long)blockIdx.x * blockDim.x + threadIdx.x;
  if (gid >= total) return;
  int c = (int)(gid % DD);
  int tok = (int)((gid / DD) % 197);
  if (tok == 0) t[gid] = cls[c] + pos[c];
  else          t[gid] += pos[(long long)tok * DD + c];
}

// ---------------------------------------------------------------- layernorm
__global__ void layernorm_f16(const float* __restrict__ in,
                              const float* __restrict__ g,
                              const float* __restrict__ b,
                              _Float16* __restrict__ out,
                              int rows, int Lout, int Lfull, int tokOff) {
  int warp = (int)(((long long)blockIdx.x * blockDim.x + threadIdx.x) >> 5);
  int lane = threadIdx.x & 31;
  if (warp >= rows) return;
  int bi = warp / Lout, l = warp % Lout;
  const float* x = in + ((long long)bi * Lfull + tokOff + l) * DD;
  float s = 0.f;
  for (int i = lane; i < DD; i += 32) s += x[i];
  for (int m = 16; m > 0; m >>= 1) s += __shfl_xor(s, m, 32);
  float mean = s / DD;
  float v = 0.f;
  for (int i = lane; i < DD; i += 32) { float d = x[i] - mean; v += d * d; }
  for (int m = 16; m > 0; m >>= 1) v += __shfl_xor(v, m, 32);
  float rstd = rsqrtf(v / DD + 1e-6f);
  _Float16* o = out + (long long)warp * DD;
  for (int i = lane; i < DD; i += 32)
    o[i] = (_Float16)((x[i] - mean) * rstd * g[i] + b[i]);
}

// ---------------------------------------------------------------- attention
__global__ void qkv_split(const float* __restrict__ qkv,
                          _Float16* __restrict__ q, _Float16* __restrict__ k,
                          _Float16* __restrict__ vT,
                          long long total, int Ntok, int Npad, float scale) {
  long long gid = (long long)blockIdx.x * blockDim.x + threadIdx.x;
  if (gid >= total) return;
  int d = (int)(gid & 63);
  long long r = gid >> 6;
  int tok = (int)(r % Npad);
  long long bh = r / Npad;
  int h = (int)(bh % NHH), b = (int)(bh / NHH);
  _Float16 qv = (_Float16)0.f, kv = (_Float16)0.f, vv = (_Float16)0.f;
  if (tok < Ntok) {
    const float* p = qkv + ((long long)(b * Ntok + tok)) * (3 * DD) + h * HDD + d;
    qv = (_Float16)(p[0] * scale);
    kv = (_Float16)(p[DD]);
    vv = (_Float16)(p[2 * DD]);
  }
  q[gid] = qv;
  k[gid] = kv;
  vT[(bh * HDD + d) * Npad + tok] = vv;
}

__global__ void softmax_rows(const float* __restrict__ sc,
                             _Float16* __restrict__ attn,
                             int rows, int Npad, int Ntok) {
  int warp = (int)(((long long)blockIdx.x * blockDim.x + threadIdx.x) >> 5);
  int lane = threadIdx.x & 31;
  if (warp >= rows) return;
  int i = warp % Npad;
  const float* srow = sc + (long long)warp * Npad;
  _Float16* arow = attn + (long long)warp * Npad;
  if (i >= Ntok) {
    for (int j = lane; j < Npad; j += 32) arow[j] = (_Float16)0.f;
    return;
  }
  float mx = -INFINITY;
  for (int j = lane; j < Ntok; j += 32) mx = fmaxf(mx, srow[j]);
  for (int m = 16; m > 0; m >>= 1) mx = fmaxf(mx, __shfl_xor(mx, m, 32));
  float sum = 0.f;
  for (int j = lane; j < Ntok; j += 32) sum += expf(srow[j] - mx);
  for (int m = 16; m > 0; m >>= 1) sum += __shfl_xor(sum, m, 32);
  float inv = 1.f / sum;
  for (int j = lane; j < Npad; j += 32)
    arow[j] = (j < Ntok) ? (_Float16)(expf(srow[j] - mx) * inv) : (_Float16)0.f;
}

__global__ void o_permute(const _Float16* __restrict__ o_tmp,
                          _Float16* __restrict__ oCat,
                          long long total, int Ntok, int Npad) {
  long long gid = (long long)blockIdx.x * blockDim.x + threadIdx.x;
  if (gid >= total) return;
  int c = (int)(gid % DD);
  long long row = gid / DD;
  int b = (int)(row / Ntok), t = (int)(row % Ntok);
  int h = c >> 6, d = c & 63;
  oCat[gid] = o_tmp[(((long long)(b * NHH + h)) * Npad + t) * HDD + d];
}

// ---------------------------------------------------------------- predictor
__global__ void pool_mean(const float* __restrict__ h1f,
                          _Float16* __restrict__ gmean, int L) {
  int id = blockIdx.x * blockDim.x + threadIdx.x;
  if (id >= BB * (DD / 2)) return;
  int b = id / (DD / 2), c = id % (DD / 2);
  float s = 0.f;
  for (int l = 0; l < L; ++l)
    s += h1f[((long long)(b * L + l)) * DD + (DD / 2) + c];
  gmean[id] = (_Float16)(s / (float)L);
}

__global__ void concat_h1(const _Float16* __restrict__ h1h,
                          const _Float16* __restrict__ gmean,
                          _Float16* __restrict__ out, long long total, int L) {
  long long gid = (long long)blockIdx.x * blockDim.x + threadIdx.x;
  if (gid >= total) return;
  int c = (int)(gid % DD);
  long long row = gid / DD;
  if (c < DD / 2) out[gid] = h1h[gid];
  else {
    int b = (int)(row / L);
    out[gid] = gmean[b * (DD / 2) + (c - DD / 2)];
  }
}

__global__ void pred_score(const _Float16* __restrict__ h3,
                           const float* __restrict__ w3,
                           const float* __restrict__ b3,
                           float* __restrict__ out, int rows) {
  int warp = (int)(((long long)blockIdx.x * blockDim.x + threadIdx.x) >> 5);
  int lane = threadIdx.x & 31;
  if (warp >= rows) return;
  const _Float16* r = h3 + (long long)warp * 192;
  float s = 0.f;
  for (int i = lane; i < 192; i += 32) s += (float)r[i] * w3[i];
  for (int m = 16; m > 0; m >>= 1) s += __shfl_xor(s, m, 32);
  if (lane == 0) {
    float x = s + b3[0];
    float ls = (x >= 0.f) ? -log1pf(expf(-x)) : (x - log1pf(expf(x)));
    out[warp] = ls;
  }
}

__global__ void topk_kernel(const float* __restrict__ scores,
                            int* __restrict__ idx, int L, int k) {
  __shared__ float sv[256];
  __shared__ float rv[256];
  __shared__ int   ri[256];
  int b = blockIdx.x, t = threadIdx.x;
  sv[t] = (t < L) ? scores[b * L + t] : -INFINITY;
  __syncthreads();
  for (int i = 0; i < k; ++i) {
    rv[t] = sv[t]; ri[t] = t;
    __syncthreads();
    for (int s = 128; s > 0; s >>= 1) {
      if (t < s) {
        float a = rv[t], c = rv[t + s];
        if (c > a || (c == a && ri[t + s] < ri[t])) { rv[t] = c; ri[t] = ri[t + s]; }
      }
      __syncthreads();
    }
    if (t == 0) { idx[b * k + i] = ri[0]; sv[ri[0]] = -INFINITY; }
    __syncthreads();
  }
}

__global__ void gather_tokens(const float* __restrict__ tin,
                              float* __restrict__ tout,
                              const int* __restrict__ idx,
                              long long total, int NtokOld, int k) {
  long long gid = (long long)blockIdx.x * blockDim.x + threadIdx.x;
  if (gid >= total) return;
  int c = (int)(gid % DD);
  long long row = gid / DD;
  int b = (int)(row / (k + 1)), to = (int)(row % (k + 1));
  int src = (to == 0) ? 0 : 1 + idx[b * k + (to - 1)];
  tout[gid] = tin[((long long)b * NtokOld + src) * DD + c];
}

// ---------------------------------------------------------------- host
static long long cdivll(long long a, long long b) { return (a + b - 1) / b; }

static void launch_gemm(hipStream_t st, const _Float16* A, long long sA,
                        const _Float16* W, long long sW, const float* bias,
                        const float* Res, long long sR, float* Cf, long long sCf,
                        _Float16* Ch, long long sCh, int M, int N, int K,
                        int act, int batch) {
  dim3 grid((unsigned)((M + 63) / 64), (unsigned)((N + 127) / 128),
            (unsigned)batch);
  gemm_wmma<<<grid, 256, 0, st>>>(A, sA, W, sW, bias, Res, sR, Cf, sCf, Ch, sCh,
                                  M, N, K, act);
}

static void launch_gemm_lds(hipStream_t st, const _Float16* A, long long sA,
                            const _Float16* W, long long sW, const float* bias,
                            const float* Res, long long sR, float* Cf, long long sCf,
                            _Float16* Ch, long long sCh, int M, int N, int K,
                            int act, int batch) {
  dim3 grid((unsigned)((M + 63) / 64), (unsigned)((N + 127) / 128),
            (unsigned)batch);
  gemm_wmma_lds<<<grid, 256, 0, st>>>(A, sA, W, sW, bias, Res, sR, Cf, sCf,
                                      Ch, sCh, M, N, K, act);
}

#define G1D(n) dim3((unsigned)cdivll((long long)(n), 256)), dim3(256), 0, stream

extern "C" void kernel_launch(void* const* d_in, const int* in_sizes, int n_in,
                              void* d_out, int out_size, void* d_ws, size_t ws_size,
                              hipStream_t stream) {
  (void)in_sizes; (void)n_in; (void)out_size; (void)ws_size;
  const float* x       = (const float*)d_in[0];
  const float* patch_w = (const float*)d_in[1];
  const float* patch_b = (const float*)d_in[2];
  const float* cls_tok = (const float*)d_in[3];
  const float* pos_emb = (const float*)d_in[4];
  const float* ln1_g   = (const float*)d_in[5];
  const float* ln1_b   = (const float*)d_in[6];
  const float* qkv_w   = (const float*)d_in[7];
  const float* qkv_b   = (const float*)d_in[8];
  const float* proj_w  = (const float*)d_in[9];
  const float* proj_b  = (const float*)d_in[10];
  const float* ln2_g   = (const float*)d_in[11];
  const float* ln2_b   = (const float*)d_in[12];
  const float* fc1_w   = (const float*)d_in[13];
  const float* fc1_b   = (const float*)d_in[14];
  const float* fc2_w   = (const float*)d_in[15];
  const float* fc2_b   = (const float*)d_in[16];
  const float* pln_g   = (const float*)d_in[17];
  const float* pln_b   = (const float*)d_in[18];
  const float* pin_w   = (const float*)d_in[19];
  const float* pin_b   = (const float*)d_in[20];
  const float* pw1     = (const float*)d_in[21];
  const float* pb1     = (const float*)d_in[22];
  const float* pw2     = (const float*)d_in[23];
  const float* pb2     = (const float*)d_in[24];
  const float* pw3     = (const float*)d_in[25];
  const float* pb3     = (const float*)d_in[26];
  const float* norm_g  = (const float*)d_in[27];
  const float* norm_b  = (const float*)d_in[28];
  const float* head_w  = (const float*)d_in[29];
  const float* head_b  = (const float*)d_in[30];

  char* base = (char*)d_ws;
  size_t off = 0;
  auto alloc = [&](size_t bytes) -> void* {
    void* p = base + off;
    off = (off + bytes + 255) & ~(size_t)255;
    return p;
  };
  const int NPM = 224;  // max padded token count (ceil32(197))

  _Float16* patch_w_h = (_Float16*)alloc(sizeof(_Float16) * 768 * 768);
  _Float16* qkv_w_h   = (_Float16*)alloc(sizeof(_Float16) * 12LL * 2304 * 768);
  _Float16* proj_w_h  = (_Float16*)alloc(sizeof(_Float16) * 12LL * 768 * 768);
  _Float16* fc1_w_h   = (_Float16*)alloc(sizeof(_Float16) * 12LL * 3072 * 768);
  _Float16* fc2_w_h   = (_Float16*)alloc(sizeof(_Float16) * 12LL * 768 * 3072);
  _Float16* pin_w_h   = (_Float16*)alloc(sizeof(_Float16) * 3LL * 768 * 768);
  _Float16* pw1_h     = (_Float16*)alloc(sizeof(_Float16) * 3LL * 384 * 768);
  _Float16* pw2_h     = (_Float16*)alloc(sizeof(_Float16) * 3LL * 192 * 384);
  _Float16* head_w_h  = (_Float16*)alloc(sizeof(_Float16) * 1000LL * 768);

  _Float16* P16   = (_Float16*)alloc(sizeof(_Float16) * (long long)BB * 196 * DD);
  float*    tA    = (float*)alloc(sizeof(float) * (long long)BB * 197 * DD);
  float*    tB    = (float*)alloc(sizeof(float) * (long long)BB * 197 * DD);
  _Float16* hbuf  = (_Float16*)alloc(sizeof(_Float16) * (long long)BB * 197 * DD);
  float*    qkvb  = (float*)alloc(sizeof(float) * (long long)BB * 197 * 3 * DD);
  _Float16* qf    = (_Float16*)alloc(sizeof(_Float16) * (long long)BB * NHH * NPM * HDD);
  _Float16* kf    = (_Float16*)alloc(sizeof(_Float16) * (long long)BB * NHH * NPM * HDD);
  _Float16* vT    = (_Float16*)alloc(sizeof(_Float16) * (long long)BB * NHH * NPM * HDD);
  float*    sc    = (float*)alloc(sizeof(float) * (long long)BB * NHH * NPM * NPM);
  _Float16* attn  = (_Float16*)alloc(sizeof(_Float16) * (long long)BB * NHH * NPM * NPM);
  _Float16* o_tmp = (_Float16*)alloc(sizeof(_Float16) * (long long)BB * NHH * NPM * HDD);
  _Float16* oCat  = (_Float16*)alloc(sizeof(_Float16) * (long long)BB * 197 * DD);
  _Float16* mid   = (_Float16*)alloc(sizeof(_Float16) * (long long)BB * 197 * 4 * DD);
  float*    h1f   = (float*)alloc(sizeof(float) * (long long)BB * 196 * DD);
  _Float16* h1h   = (_Float16*)alloc(sizeof(_Float16) * (long long)BB * 196 * DD);
  _Float16* h1cat = (_Float16*)alloc(sizeof(_Float16) * (long long)BB * 196 * DD);
  _Float16* h2b   = (_Float16*)alloc(sizeof(_Float16) * (long long)BB * 196 * 384);
  _Float16* h3b   = (_Float16*)alloc(sizeof(_Float16) * (long long)BB * 196 * 192);
  _Float16* gmean = (_Float16*)alloc(sizeof(_Float16) * (long long)BB * 384);
  float*    psc   = (float*)alloc(sizeof(float) * (long long)BB * 196);
  int*      pidx  = (int*)alloc(sizeof(int) * (long long)BB * 196);

  // --- weights to f16 (stays resident in the 192MB L2 across the pass)
  f32_to_f16<<<G1D(768LL * 768)>>>(patch_w, patch_w_h, 768LL * 768);
  f32_to_f16<<<G1D(12LL * 2304 * 768)>>>(qkv_w, qkv_w_h, 12LL * 2304 * 768);
  f32_to_f16<<<G1D(12LL * 768 * 768)>>>(proj_w, proj_w_h, 12LL * 768 * 768);
  f32_to_f16<<<G1D(12LL * 3072 * 768)>>>(fc1_w, fc1_w_h, 12LL * 3072 * 768);
  f32_to_f16<<<G1D(12LL * 768 * 3072)>>>(fc2_w, fc2_w_h, 12LL * 768 * 3072);
  f32_to_f16<<<G1D(3LL * 768 * 768)>>>(pin_w, pin_w_h, 3LL * 768 * 768);
  f32_to_f16<<<G1D(3LL * 384 * 768)>>>(pw1, pw1_h, 3LL * 384 * 768);
  f32_to_f16<<<G1D(3LL * 192 * 384)>>>(pw2, pw2_h, 3LL * 192 * 384);
  f32_to_f16<<<G1D(1000LL * 768)>>>(head_w, head_w_h, 1000LL * 768);

  // --- patch embed
  long long tim = (long long)BB * 196 * DD;
  im2col_kernel<<<G1D(tim)>>>(x, P16, tim);
  launch_gemm_lds(stream, P16, 196LL * DD, patch_w_h, 0, patch_b, nullptr, 0,
                  tA + DD, 197LL * DD, nullptr, 0, 196, DD, DD, 0, BB);
  long long tall = (long long)BB * 197 * DD;
  add_clspos<<<G1D(tall)>>>(tA, cls_tok, pos_emb, tall);

  float* tCur = tA;
  float* tNxt = tB;
  int Ntok = 197;
  int stage = 0;
  const float SCALE = 0.125f;  // 64^-0.5

  for (int i = 0; i < 12; ++i) {
    if (i == 3 || i == 6 || i == 9) {
      int L = Ntok - 1;
      int rows = BB * L;
      layernorm_f16<<<G1D((long long)rows * 32)>>>(tCur, pln_g + stage * DD,
                                                   pln_b + stage * DD, hbuf,
                                                   rows, L, Ntok, 1);
      launch_gemm_lds(stream, hbuf, 0, pin_w_h + (long long)stage * DD * DD, 0,
                      pin_b + stage * DD, nullptr, 0, h1f, 0, h1h, 0,
                      rows, DD, DD, 1, 1);
      pool_mean<<<G1D(BB * 384)>>>(h1f, gmean, L);
      concat_h1<<<G1D((long long)rows * DD)>>>(h1h, gmean, h1cat,
                                               (long long)rows * DD, L);
      launch_gemm_lds(stream, h1cat, 0, pw1_h + (long long)stage * 384 * DD, 0,
                      pb1 + stage * 384, nullptr, 0, nullptr, 0, h2b, 0,
                      rows, 384, DD, 1, 1);
      launch_gemm_lds(stream, h2b, 0, pw2_h + (long long)stage * 192 * 384, 0,
                      pb2 + stage * 192, nullptr, 0, nullptr, 0, h3b, 0,
                      rows, 192, 384, 1, 1);
      pred_score<<<G1D((long long)rows * 32)>>>(h3b, pw3 + stage * 192,
                                                pb3 + stage, psc, rows);
      int kk = (7 * L + 9) / 10;  // ceil(0.7 * L)
      topk_kernel<<<dim3(BB), dim3(256), 0, stream>>>(psc, pidx, L, kk);
      int NtokNew = kk + 1;
      gather_tokens<<<G1D((long long)BB * NtokNew * DD)>>>(
          tCur, tNxt, pidx, (long long)BB * NtokNew * DD, Ntok, kk);
      float* tmp = tCur; tCur = tNxt; tNxt = tmp;
      Ntok = NtokNew;
      ++stage;
    }

    int rows = BB * Ntok;
    int Npad = (Ntok + 31) & ~31;

    layernorm_f16<<<G1D((long long)rows * 32)>>>(tCur, ln1_g + i * DD,
                                                 ln1_b + i * DD, hbuf,
                                                 rows, Ntok, Ntok, 0);
    launch_gemm_lds(stream, hbuf, 0, qkv_w_h + (long long)i * 2304 * DD, 0,
                    qkv_b + i * 2304, nullptr, 0, qkvb, 0, nullptr, 0,
                    rows, 2304, DD, 0, 1);
    long long tsp = (long long)BB * NHH * Npad * HDD;
    qkv_split<<<G1D(tsp)>>>(qkvb, qf, kf, vT, tsp, Ntok, Npad, SCALE);
    launch_gemm(stream, qf, (long long)Npad * HDD, kf, (long long)Npad * HDD,
                nullptr, nullptr, 0, sc, (long long)Npad * Npad, nullptr, 0,
                Npad, Npad, HDD, 0, BB * NHH);
    int srows = BB * NHH * Npad;
    softmax_rows<<<G1D((long long)srows * 32)>>>(sc, attn, srows, Npad, Ntok);
    launch_gemm(stream, attn, (long long)Npad * Npad, vT, (long long)HDD * Npad,
                nullptr, nullptr, 0, nullptr, 0, o_tmp, (long long)Npad * HDD,
                Ntok, HDD, Npad, 0, BB * NHH);
    o_permute<<<G1D((long long)rows * DD)>>>(o_tmp, oCat, (long long)rows * DD,
                                             Ntok, Npad);
    launch_gemm_lds(stream, oCat, 0, proj_w_h + (long long)i * DD * DD, 0,
                    proj_b + i * DD, tCur, 0, tCur, 0, nullptr, 0,
                    rows, DD, DD, 0, 1);
    layernorm_f16<<<G1D((long long)rows * 32)>>>(tCur, ln2_g + i * DD,
                                                 ln2_b + i * DD, hbuf,
                                                 rows, Ntok, Ntok, 0);
    launch_gemm_lds(stream, hbuf, 0, fc1_w_h + (long long)i * 3072 * DD, 0,
                    fc1_b + i * 3072, nullptr, 0, nullptr, 0, mid, 0,
                    rows, 3072, DD, 1, 1);
    launch_gemm_lds(stream, mid, 0, fc2_w_h + (long long)i * DD * 3072, 0,
                    fc2_b + i * DD, tCur, 0, tCur, 0, nullptr, 0,
                    rows, DD, 3072, 0, 1);
  }

  // final LN on CLS token only, then head
  layernorm_f16<<<G1D((long long)BB * 32)>>>(tCur, norm_g, norm_b, hbuf,
                                             BB, 1, Ntok, 0);
  launch_gemm_lds(stream, hbuf, 0, head_w_h, 0, head_b, nullptr, 0,
                  (float*)d_out, 0, nullptr, 0, BB, 1000, DD, 0, 1);
}
